// MultiHeadAttention_31756988187106
// MI455X (gfx1250) — compile-verified, matrix operand order checked by reading, not run on hardware
//
#include <hip/hip_runtime.h>

typedef _Float16 f16;
typedef _Float16 v16h __attribute__((ext_vector_type(16)));
typedef _Float16 v8h  __attribute__((ext_vector_type(8)));
typedef float    v8f  __attribute__((ext_vector_type(8)));
typedef float    v4f  __attribute__((ext_vector_type(4)));

#define WMMA_F32_F16(a,b,c) \
  __builtin_amdgcn_wmma_f32_16x16x32_f16(false,(a),false,(b),(short)0,(c),false,false)

static constexpr int H  = 8;
static constexpr int DH = 64;
static constexpr int DM = 512;   // H*DH
static constexpr int T  = 2048;

__device__ __forceinline__ v16h mk16(v8h lo, v8h hi){
  return __builtin_shufflevector(lo, hi, 0,1,2,3,4,5,6,7,8,9,10,11,12,13,14,15);
}

// Fragment loader for f16 row-major source (serves both A and B operands).
// Lane L holds row/col (row0 + (L&15)); element j maps to
// k = kc + (j/8)*16 + (L>>4)*8 + (j%8)  -> two contiguous 16B loads.
__device__ __forceinline__ v16h frag_f16(const f16* p, int stride, int row0, int kc, int lane){
  const int l16 = lane & 15, hl = (lane >> 4) & 1;
  const f16* q = p + (size_t)(row0 + l16) * stride + kc + hl * 8;
  v8h lo = *(const v8h*)(q);
  v8h hi = *(const v8h*)(q + 16);
  return mk16(lo, hi);
}

// Same mapping, f32 source with on-the-fly convert to f16.
__device__ __forceinline__ v16h frag_f32(const float* p, int stride, int row0, int kc, int lane){
  const int l16 = lane & 15, hl = (lane >> 4) & 1;
  const float* q = p + (size_t)(row0 + l16) * stride + kc + hl * 8;
  v4f a0 = *(const v4f*)(q);
  v4f a1 = *(const v4f*)(q + 4);
  v4f b0 = *(const v4f*)(q + 16);
  v4f b1 = *(const v4f*)(q + 20);
  v16h r;
  #pragma unroll
  for (int i = 0; i < 4; ++i){
    r[i]      = (f16)a0[i];
    r[4 + i]  = (f16)a1[i];
    r[8 + i]  = (f16)b0[i];
    r[12 + i] = (f16)b1[i];
  }
  return r;
}

// ---------------- kernel 0: Wo f32 -> f16 ----------------
__global__ void __launch_bounds__(256) cvt_kernel(const float* __restrict__ src,
                                                  f16* __restrict__ dst, int n){
  for (int i = blockIdx.x * blockDim.x + threadIdx.x; i < n; i += gridDim.x * blockDim.x)
    dst[i] = (f16)src[i];
}

// ---------------- kernel 1: per-head QKV projection ----------------
// One wave computes a 16(time) x 64(dim) tile for Q, K and V of one (b,h).
// Q stored row-major pre-scaled by 1/8; K row-major; V stored transposed [bh][d][T].
__global__ void __launch_bounds__(128) qkv_proj_kernel(
    const float* __restrict__ qi, const float* __restrict__ ki, const float* __restrict__ vi,
    const float* __restrict__ Wq, const float* __restrict__ bq,
    const float* __restrict__ Wk, const float* __restrict__ bk,
    const float* __restrict__ Wv, const float* __restrict__ bv,
    f16* __restrict__ Qh, f16* __restrict__ Kh, f16* __restrict__ Vt, int B)
{
  const int lane = threadIdx.x & 31;
  const int w    = threadIdx.x >> 5;
  const int gt   = blockIdx.x * 4 + w;
  const int nTq  = T / 16;
  const int bh   = gt / nTq;
  if (bh >= B * H) return;
  const int t0   = (gt % nTq) * 16;
  const int b = bh / H, h = bh % H;
  const int l16 = lane & 15, hl = (lane >> 4) & 1;

  const float* srcs[3] = { qi, ki, vi };
  const float* Ws[3]   = { Wq, Wk, Wv };
  const float* bs[3]   = { bq, bk, bv };

  #pragma unroll
  for (int p = 0; p < 3; ++p) {
    const float* src = srcs[p] + (size_t)b * T * DM + h * DH;
    v16h a0 = frag_f32(src, DM, t0, 0,  lane);   // rows t0.., k = head dim 0..31
    v16h a1 = frag_f32(src, DM, t0, 32, lane);   // k = 32..63
    #pragma unroll
    for (int nc = 0; nc < 4; ++nc) {
      // B[k][n] = W[n][k]  (torch x @ W.T): column n is W row n, contiguous in k.
      v16h b0 = frag_f32(Ws[p], DH, nc * 16, 0,  lane);
      v16h b1 = frag_f32(Ws[p], DH, nc * 16, 32, lane);
      v8f acc = {};
      acc = WMMA_F32_F16(a0, b0, acc);
      acc = WMMA_F32_F16(a1, b1, acc);
      const float bias = bs[p][nc * 16 + l16];
      #pragma unroll
      for (int r = 0; r < 8; ++r) {
        const int m = r + 8 * hl;                // time row within tile
        const float val = acc[r] + bias;
        if (p == 0) {
          Qh[((size_t)bh * T + t0 + m) * DH + nc * 16 + l16] = (f16)(val * 0.125f);
        } else if (p == 1) {
          Kh[((size_t)bh * T + t0 + m) * DH + nc * 16 + l16] = (f16)val;
        } else {
          Vt[((size_t)bh * DH + nc * 16 + l16) * T + t0 + m] = (f16)val;
        }
      }
    }
  }
}

// ---------------- kernel 2: flash attention ----------------
// One wave owns 16 query rows of one (b,h); streams keys in chunks of 32.
__global__ void __launch_bounds__(256) attn_kernel(
    const f16* __restrict__ Qh, const f16* __restrict__ Kh, const f16* __restrict__ Vt,
    f16* __restrict__ Ch, int B)
{
  __shared__ __align__(16) f16 plds[8 * 16 * 32];   // 1KB staging tile per wave
  const int lane = threadIdx.x & 31;
  const int w    = threadIdx.x >> 5;
  const int gt   = blockIdx.x * 8 + w;
  const int nTq  = T / 16;
  const int bh   = gt / nTq;
  if (bh >= B * H) return;
  const int t0   = (gt % nTq) * 16;
  const int b = bh / H, h = bh % H;
  const int l16 = lane & 15, hl = (lane >> 4) & 1;
  f16* myP = plds + w * (16 * 32);

  const f16* Qb = Qh + (size_t)bh * T * DH;
  const f16* Kb = Kh + (size_t)bh * T * DH;
  const f16* Vb = Vt + (size_t)bh * DH * T;

  // Q tile A-fragments, resident for the whole sweep (pre-scaled by 1/8)
  const v16h qa0 = frag_f16(Qb, DH, t0, 0,  lane);
  const v16h qa1 = frag_f16(Qb, DH, t0, 32, lane);

  float mrow[8], lrow[8];
  v8f o0 = {}, o1 = {}, o2 = {}, o3 = {};
  #pragma unroll
  for (int r = 0; r < 8; ++r) { mrow[r] = -3.0e38f; lrow[r] = 0.f; }

  #pragma unroll 1
  for (int kv = 0; kv < T; kv += 32) {
    if (kv + 32 < T)   // hint next K chunk toward the caches (global_prefetch_b8)
      __builtin_prefetch(Kb + (size_t)(kv + 32) * DH + lane * 4, 0, 3);

    // ---- scores: two 16-key tiles, K-dim = 64 split into 2 WMMAs each
    v8f s0 = {}, s1 = {};
    {
      v16h kb0 = frag_f16(Kb, DH, kv,      0,  lane);  // B[k=d][n=key]: K row per key
      v16h kb1 = frag_f16(Kb, DH, kv,      32, lane);
      s0 = WMMA_F32_F16(qa0, kb0, s0);
      s0 = WMMA_F32_F16(qa1, kb1, s0);
      v16h kb2 = frag_f16(Kb, DH, kv + 16, 0,  lane);
      v16h kb3 = frag_f16(Kb, DH, kv + 16, 32, lane);
      s1 = WMMA_F32_F16(qa0, kb2, s1);
      s1 = WMMA_F32_F16(qa1, kb3, s1);
    }

    // ---- online softmax: rows live across the 16 lanes of each half-wave
    float alpha[8];
    #pragma unroll
    for (int r = 0; r < 8; ++r) {
      float mx = fmaxf(s0[r], s1[r]);
      #pragma unroll
      for (int off = 8; off >= 1; off >>= 1) mx = fmaxf(mx, __shfl_xor(mx, off, 16));
      const float mn = fmaxf(mrow[r], mx);
      const float a  = __expf(mrow[r] - mn);
      const float p0 = __expf(s0[r] - mn);
      const float p1 = __expf(s1[r] - mn);
      float sm = p0 + p1;
      #pragma unroll
      for (int off = 8; off >= 1; off >>= 1) sm += __shfl_xor(sm, off, 16);
      lrow[r] = lrow[r] * a + sm;
      mrow[r] = mn;
      alpha[r] = a;
      s0[r] = p0; s1[r] = p1;
    }
    #pragma unroll
    for (int r = 0; r < 8; ++r) {
      o0[r] *= alpha[r]; o1[r] *= alpha[r]; o2[r] *= alpha[r]; o3[r] *= alpha[r];
    }

    // ---- D-layout -> A-layout for P via wave-private LDS (in-order DS per wave)
    #pragma unroll
    for (int r = 0; r < 8; ++r) {
      const int m = r + 8 * hl;
      myP[m * 32 + l16]      = (f16)s0[r];
      myP[m * 32 + 16 + l16] = (f16)s1[r];
    }
    __builtin_amdgcn_wave_barrier();
    const v16h pa = frag_f16(myP, 32, 0, 0, lane);

    // ---- ctx += P(16x32) x V(32x64); Vt gives contiguous per-lane key runs
    v16h vb0 = frag_f16(Vb, T, 0,  kv, lane);
    v16h vb1 = frag_f16(Vb, T, 16, kv, lane);
    v16h vb2 = frag_f16(Vb, T, 32, kv, lane);
    v16h vb3 = frag_f16(Vb, T, 48, kv, lane);
    o0 = WMMA_F32_F16(pa, vb0, o0);
    o1 = WMMA_F32_F16(pa, vb1, o1);
    o2 = WMMA_F32_F16(pa, vb2, o2);
    o3 = WMMA_F32_F16(pa, vb3, o3);
    __builtin_amdgcn_wave_barrier();
  }

  // ---- normalize and emit concat layout [b][t][h*64 + d] as f16
  #pragma unroll
  for (int r = 0; r < 8; ++r) {
    const float rl = (lrow[r] > 0.f) ? (1.0f / lrow[r]) : 0.f;
    const int m = r + 8 * hl;
    const size_t base = ((size_t)b * T + t0 + m) * DM + h * DH + l16;
    Ch[base +  0] = (f16)(o0[r] * rl);
    Ch[base + 16] = (f16)(o1[r] * rl);
    Ch[base + 32] = (f16)(o2[r] * rl);
    Ch[base + 48] = (f16)(o3[r] * rl);
  }
}

// ---------------- kernel 3: output projection [B*T,512] x Wo^T + bo ----------------
__global__ void __launch_bounds__(256) out_proj_kernel(
    const f16* __restrict__ Ch, const f16* __restrict__ Woh,
    const float* __restrict__ bo, float* __restrict__ out, int B)
{
  const int lane = threadIdx.x & 31;
  const int w    = threadIdx.x >> 5;
  const int gt   = blockIdx.x * 8 + w;
  const int nN   = DM / 64;            // 8 column groups
  const int mt   = gt / nN;
  const int M    = B * T;
  if (mt * 16 >= M) return;
  const int n0   = (gt % nN) * 64;
  const int m0   = mt * 16;
  const int l16 = lane & 15, hl = (lane >> 4) & 1;

  v8f acc0 = {}, acc1 = {}, acc2 = {}, acc3 = {};
  #pragma unroll 2
  for (int kc = 0; kc < DM; kc += 32) {
    v16h a  = frag_f16(Ch,  DM, m0,      kc, lane);
    v16h b0 = frag_f16(Woh, DM, n0,      kc, lane);  // B[k][n] = Wo[n][k]
    v16h b1 = frag_f16(Woh, DM, n0 + 16, kc, lane);
    v16h b2 = frag_f16(Woh, DM, n0 + 32, kc, lane);
    v16h b3 = frag_f16(Woh, DM, n0 + 48, kc, lane);
    acc0 = WMMA_F32_F16(a, b0, acc0);
    acc1 = WMMA_F32_F16(a, b1, acc1);
    acc2 = WMMA_F32_F16(a, b2, acc2);
    acc3 = WMMA_F32_F16(a, b3, acc3);
  }
  #pragma unroll
  for (int r = 0; r < 8; ++r) {
    const int m = m0 + r + 8 * hl;
    float* op = out + (size_t)m * DM + n0;
    op[l16]      = acc0[r] + bo[n0 +      l16];
    op[16 + l16] = acc1[r] + bo[n0 + 16 + l16];
    op[32 + l16] = acc2[r] + bo[n0 + 32 + l16];
    op[48 + l16] = acc3[r] + bo[n0 + 48 + l16];
  }
}

extern "C" void kernel_launch(void* const* d_in, const int* in_sizes, int n_in,
                              void* d_out, int out_size, void* d_ws, size_t ws_size,
                              hipStream_t stream)
{
  const float* q  = (const float*)d_in[0];
  const float* k  = (const float*)d_in[1];
  const float* v  = (const float*)d_in[2];
  const float* Wq = (const float*)d_in[3];
  const float* bq = (const float*)d_in[4];
  const float* Wk = (const float*)d_in[5];
  const float* bk = (const float*)d_in[6];
  const float* Wv = (const float*)d_in[7];
  const float* bv = (const float*)d_in[8];
  const float* Wo = (const float*)d_in[9];
  const float* bo = (const float*)d_in[10];
  float* out = (float*)d_out;

  const int B = in_sizes[0] / (T * DM);

  // Workspace partition (f16): Q | K | Vt | ctx | Wo  ->  ~32.5 MB for B=4
  const size_t headSz = (size_t)B * H * T * DH;
  f16* Qh  = (f16*)d_ws;
  f16* Kh  = Qh + headSz;
  f16* Vt  = Kh + headSz;
  f16* Ch  = Vt + headSz;
  f16* Woh = Ch + (size_t)B * T * DM;

  cvt_kernel<<<dim3(128), dim3(256), 0, stream>>>(Wo, Woh, DM * DM);

  const int tiles = B * H * (T / 16);
  qkv_proj_kernel<<<dim3((tiles + 3) / 4), dim3(128), 0, stream>>>(
      q, k, v, Wq, bq, Wk, bk, Wv, bv, Qh, Kh, Vt, B);

  attn_kernel<<<dim3((tiles + 7) / 8), dim3(256), 0, stream>>>(Qh, Kh, Vt, Ch, B);

  const int wt3 = (B * T / 16) * (DM / 64);
  out_proj_kernel<<<dim3((wt3 + 7) / 8), dim3(256), 0, stream>>>(Ch, Woh, bo, out, B);
}
